// GCViTBlock_27384711479403
// MI455X (gfx1250) — matrix operand order repaired and checked by reading, not compiled, not run on hardware
//
#include <hip/hip_runtime.h>
#include <hip/hip_bf16.h>

typedef __bf16 bf16_t;
typedef __attribute__((ext_vector_type(16))) __bf16 v16bf;
typedef __attribute__((ext_vector_type(8)))  float  v8f;

union AFrag { v16bf v; uint4 q[2]; };

__device__ __forceinline__ v8f wmma_bf16(v16bf a, v16bf b, v8f c) {
  return __builtin_amdgcn_wmma_f32_16x16x32_bf16(false, a, false, b, (short)0, c, false, false);
}

// ---------------------------------------------------------------------------
// small prep kernels
// ---------------------------------------------------------------------------
__global__ void cvt_f32_bf16(const float* __restrict__ in, bf16_t* __restrict__ out, int n) {
  int i = blockIdx.x * blockDim.x + threadIdx.x;
  if (i < n) out[i] = (bf16_t)in[i];
}

// Padded bias (8,64,64): rel-pos bias for (q<49,k<49); 0 for padded queries;
// -1e30 for padded keys (softmax mask baked in -> branch-free attention).
__global__ void build_bias_pad(const float* __restrict__ rpb, float* __restrict__ biasP) {
  int idx = blockIdx.x * blockDim.x + threadIdx.x;
  if (idx >= 8 * 64 * 64) return;
  int head = idx >> 12, r = idx & 4095;
  int q = r >> 6, k = r & 63;
  float v;
  if (k >= 49) {
    v = -1e30f;
  } else if (q >= 49) {
    v = 0.f;
  } else {
    int qi = q / 7, qj = q % 7, ki = k / 7, kj = k % 7;
    int rel = (qi - ki + 6) * 13 + (qj - kj + 6);
    v = rpb[rel * 8 + head];
  }
  biasP[idx] = v;
}

// LayerNorm over C=256, wave-per-row (8 rows / 256-thread block), bf16 out.
// win_order=1: scatter rows into window-partition order for the attention path.
__global__ __launch_bounds__(256)
void ln_rows(const float* __restrict__ x, const float* __restrict__ g,
             const float* __restrict__ b, bf16_t* __restrict__ out, int win_order) {
  const int wid = threadIdx.x >> 5, lane = threadIdx.x & 31;
  const size_t row = (size_t)blockIdx.x * 8 + wid;
  const float* xr = x + row * 256 + lane * 8;
  float4 v0 = ((const float4*)xr)[0];
  float4 v1 = ((const float4*)xr)[1];
  float vv[8] = {v0.x, v0.y, v0.z, v0.w, v1.x, v1.y, v1.z, v1.w};
  float s = 0.f, s2 = 0.f;
#pragma unroll
  for (int i = 0; i < 8; ++i) { s += vv[i]; s2 += vv[i] * vv[i]; }
#pragma unroll
  for (int m = 16; m >= 1; m >>= 1) {
    s  += __shfl_xor(s, m, 32);
    s2 += __shfl_xor(s2, m, 32);
  }
  const float mean = s * (1.f / 256.f);
  const float var  = s2 * (1.f / 256.f) - mean * mean;
  const float rstd = rsqrtf(var + 1e-5f);
  size_t orow = row;
  if (win_order) {
    int pix = (int)row;
    int b_ = pix / 3136, rem = pix % 3136;
    int hh = rem / 56, ww = rem % 56;
    int wy = hh / 7, ii = hh % 7, wx = ww / 7, jj = ww % 7;
    orow = (size_t)((b_ * 64 + wy * 8 + wx) * 49 + ii * 7 + jj);
  }
  const int c0 = lane * 8;
  float4 g0 = ((const float4*)(g + c0))[0];
  float4 g1 = ((const float4*)(g + c0))[1];
  float4 b0 = ((const float4*)(b + c0))[0];
  float4 b1 = ((const float4*)(b + c0))[1];
  float gg[8] = {g0.x, g0.y, g0.z, g0.w, g1.x, g1.y, g1.z, g1.w};
  float bb[8] = {b0.x, b0.y, b0.z, b0.w, b1.x, b1.y, b1.z, b1.w};
  union { bf16_t h[8]; uint4 q; } o;
#pragma unroll
  for (int i = 0; i < 8; ++i) {
    float y = (vv[i] - mean) * rstd * gg[i] + bb[i];
    o.h[i] = (bf16_t)y;
  }
  *(uint4*)(out + orow * 256 + c0) = o.q;
}

// ---------------------------------------------------------------------------
// Tiled bf16 WMMA GEMM: out[M,Nout] = A[M,K] @ W[Nout,K]^T (+epilogue)
// Block tile 128(M) x 128(N), K-step 32. 8 waves as 4(M) x 2(N):
// each wave computes 32x64 via 8 WMMAs/K-step (2 A-frags x 4 B-frags).
// ---------------------------------------------------------------------------
constexpr int EPI_KV = 0, EPI_PROJ = 1, EPI_FC1 = 2, EPI_FC2 = 3;
#define TM 128
#define TN 128
#define TK 32
#define LDT 40   // padded LDS row stride (halves): 80B -> conflict-free b128 reads

template <int EPI>
__global__ __launch_bounds__(256)
void gemm_bf16_ep(const bf16_t* __restrict__ A, const bf16_t* __restrict__ W,
                  const float* __restrict__ bias, const float* __restrict__ resid,
                  float* __restrict__ outF, bf16_t* __restrict__ outB,
                  int K, int Nout) {
  __shared__ bf16_t As[TM * LDT];
  __shared__ bf16_t Bs[TN * LDT];
  const int tid  = threadIdx.x;
  const int lane = tid & 31, wid = tid >> 5;
  const int half = lane >> 4, l15 = lane & 15;
  const int m0g = blockIdx.x * TM;
  const int n0g = blockIdx.y * TN;
  const int m0w = (wid & 3) * 32;   // wave M offset within block tile
  const int n0w = (wid >> 2) * 64;  // wave N offset within block tile

  v8f acc[2][4];
#pragma unroll
  for (int mt = 0; mt < 2; ++mt)
#pragma unroll
    for (int nt = 0; nt < 4; ++nt)
#pragma unroll
      for (int r = 0; r < 8; ++r) acc[mt][nt][r] = 0.f;

  const int sr = tid >> 1, sc = (tid & 1) * 16;  // staging: 16 halves/thread each

  for (int k0 = 0; k0 < K; k0 += TK) {
    const uint4* ga = (const uint4*)(A + (size_t)(m0g + sr) * K + k0 + sc);
    uint4 a0 = ga[0], a1 = ga[1];
    const uint4* gb = (const uint4*)(W + (size_t)(n0g + sr) * K + k0 + sc);
    uint4 b0 = gb[0], b1 = gb[1];
    __syncthreads();
    *(uint4*)(As + sr * LDT + sc)     = a0;
    *(uint4*)(As + sr * LDT + sc + 8) = a1;
    *(uint4*)(Bs + sr * LDT + sc)     = b0;
    *(uint4*)(Bs + sr * LDT + sc + 8) = b1;
    __syncthreads();
    // A fragments (ISA 16-bit A layout: lane half selects K 0-7/16-23 vs 8-15/24-31)
    AFrag af[2];
#pragma unroll
    for (int mt = 0; mt < 2; ++mt) {
      const bf16_t* ap = As + (m0w + mt * 16 + l15) * LDT + half * 8;
      af[mt].q[0] = *(const uint4*)ap;
      af[mt].q[1] = *(const uint4*)(ap + 16);
    }
    // B fragments: preload all 4 into distinct registers so the DS loads
    // can retire under staggered waits while WMMAs issue.
    AFrag bfg[4];
#pragma unroll
    for (int nt = 0; nt < 4; ++nt) {
      const bf16_t* bp = Bs + (n0w + nt * 16 + l15) * LDT + half * 8;
      bfg[nt].q[0] = *(const uint4*)bp;
      bfg[nt].q[1] = *(const uint4*)(bp + 16);
    }
#pragma unroll
    for (int nt = 0; nt < 4; ++nt) {
      acc[0][nt] = wmma_bf16(af[0].v, bfg[nt].v, acc[0][nt]);
      acc[1][nt] = wmma_bf16(af[1].v, bfg[nt].v, acc[1][nt]);
    }
  }

  // Epilogue. C layout: VGPR r, lanes<16 -> M=r, lanes>=16 -> M=8+r; N = lane&15.
#pragma unroll
  for (int mt = 0; mt < 2; ++mt) {
#pragma unroll
    for (int r = 0; r < 8; ++r) {
      const int row = m0g + m0w + mt * 16 + 8 * half + r;
      size_t pix = 0;
      if (EPI == EPI_PROJ) {  // window-reverse mapping row -> raster pixel
        int win = row / 49, n = row % 49;
        int b_ = win >> 6, w_ = win & 63;
        int wy = w_ >> 3, wx = w_ & 7;
        int ii = n / 7, jj = n % 7;
        pix = (size_t)b_ * 3136 + (wy * 7 + ii) * 56 + (wx * 7 + jj);
      }
#pragma unroll
      for (int nt = 0; nt < 4; ++nt) {
        const int col = n0g + n0w + nt * 16 + l15;
        float v = acc[mt][nt][r] + bias[col];
        if (EPI == EPI_KV) {
          outB[(size_t)row * Nout + col] = (bf16_t)v;
        } else if (EPI == EPI_PROJ) {
          outF[pix * 256 + col] = resid[pix * 256 + col] + v;
        } else if (EPI == EPI_FC1) {
          float ge = 0.5f * v * (1.0f + erff(v * 0.70710678118f));
          outB[(size_t)row * Nout + col] = (bf16_t)ge;
        } else {  // EPI_FC2
          outF[(size_t)row * 256 + col] = resid[(size_t)row * 256 + col] + v;
        }
      }
    }
  }
}

// ---------------------------------------------------------------------------
// Windowed attention: one block per (window, head). 4 waves, N=49 padded to 64.
// S = (q*scale) @ k^T + biasP -> softmax -> P @ v. All matmuls via bf16 WMMA.
// biasP is (8,64,64) with key-mask baked in, so this kernel is branch-free.
// ---------------------------------------------------------------------------
__global__ __launch_bounds__(128)
void attn_win(const bf16_t* __restrict__ qb, const bf16_t* __restrict__ kvb,
              const float* __restrict__ biasP, bf16_t* __restrict__ o) {
  __shared__ bf16_t qs[64 * LDT];
  __shared__ bf16_t ks[64 * LDT];
  __shared__ bf16_t vts[32 * 72];  // V transposed: vts[d][key]
  __shared__ bf16_t ps[64 * 72];   // softmax probabilities
  const int bh = blockIdx.x;
  const int win = bh >> 3, head = bh & 7;
  const int b_ = win >> 6;
  const int tid = threadIdx.x, lane = tid & 31, wid = tid >> 5;
  const int half = lane >> 4, l15 = lane & 15;
  const int m0 = wid * 16;

  // stage q, k (zero-padded rows 49..63) and v (transposed)
  {
    const int row = tid >> 1, c = tid & 1;
    const uint4 z = {0u, 0u, 0u, 0u};
    uint4 q0 = z, q1 = z, k0 = z, k1 = z;
    union { bf16_t h[16]; uint4 q[2]; } vt;
    vt.q[0] = z; vt.q[1] = z;
    if (row < 49) {
      const bf16_t* sq = qb + ((size_t)(b_ * 8 + head) * 49 + row) * 32 + c * 16;
      q0 = ((const uint4*)sq)[0];
      q1 = ((const uint4*)sq)[1];
      const bf16_t* sk = kvb + ((size_t)win * 49 + row) * 512 + head * 32 + c * 16;
      k0 = ((const uint4*)sk)[0];
      k1 = ((const uint4*)sk)[1];
      const bf16_t* sv = kvb + ((size_t)win * 49 + row) * 512 + 256 + head * 32 + c * 16;
      vt.q[0] = ((const uint4*)sv)[0];
      vt.q[1] = ((const uint4*)sv)[1];
    }
    *(uint4*)(qs + row * LDT + c * 16)     = q0;
    *(uint4*)(qs + row * LDT + c * 16 + 8) = q1;
    *(uint4*)(ks + row * LDT + c * 16)     = k0;
    *(uint4*)(ks + row * LDT + c * 16 + 8) = k1;
#pragma unroll
    for (int i = 0; i < 16; ++i) vts[(c * 16 + i) * 72 + row] = vt.h[i];
  }
  __syncthreads();

  // S = q @ k^T : wave computes 16 query rows x 64 keys, K=32 one WMMA step.
  // Preload all B fragments (distinct regs) -> pipelined ds_load/wmma.
  v8f acc[4];
#pragma unroll
  for (int nt = 0; nt < 4; ++nt)
#pragma unroll
    for (int r = 0; r < 8; ++r) acc[nt][r] = 0.f;
  {
    AFrag af;
    const bf16_t* ap = qs + (m0 + l15) * LDT + half * 8;
    af.q[0] = *(const uint4*)ap;
    af.q[1] = *(const uint4*)(ap + 16);
    AFrag bfg[4];
#pragma unroll
    for (int nt = 0; nt < 4; ++nt) {
      const bf16_t* bp = ks + (nt * 16 + l15) * LDT + half * 8;
      bfg[nt].q[0] = *(const uint4*)bp;
      bfg[nt].q[1] = *(const uint4*)(bp + 16);
    }
#pragma unroll
    for (int nt = 0; nt < 4; ++nt) {
      acc[nt] = wmma_bf16(af.v, bfg[nt].v, acc[nt]);
    }
  }

  // scale + padded bias (mask baked in) + softmax, write P (bf16) to LDS
  const float scale = 0.17677669529f;  // 1/sqrt(32)
  const float* brow = biasP + (head << 12);
#pragma unroll
  for (int r = 0; r < 8; ++r) {
    const int qrow = m0 + 8 * half + r;
    float s0[4];
    float mx = -1e30f;
#pragma unroll
    for (int nt = 0; nt < 4; ++nt) {
      const int key = nt * 16 + l15;
      float v = acc[nt][r] * scale + brow[(qrow << 6) + key];
      s0[nt] = v;
      mx = fmaxf(mx, v);
    }
#pragma unroll
    for (int m = 8; m >= 1; m >>= 1) mx = fmaxf(mx, __shfl_xor(mx, m, 32));
    float sum = 0.f;
#pragma unroll
    for (int nt = 0; nt < 4; ++nt) { s0[nt] = __expf(s0[nt] - mx); sum += s0[nt]; }
#pragma unroll
    for (int m = 8; m >= 1; m >>= 1) sum += __shfl_xor(sum, m, 32);
    const float inv = 1.0f / sum;
#pragma unroll
    for (int nt = 0; nt < 4; ++nt) {
      const int key = nt * 16 + l15;
      ps[qrow * 72 + key] = (bf16_t)(s0[nt] * inv);
    }
  }
  __syncthreads();

  // O = P @ V : K=64 (2 steps), N=32 (2 subtiles); preload fragments per step.
  v8f oacc[2];
#pragma unroll
  for (int nt = 0; nt < 2; ++nt)
#pragma unroll
    for (int r = 0; r < 8; ++r) oacc[nt][r] = 0.f;
#pragma unroll
  for (int kstep = 0; kstep < 2; ++kstep) {
    AFrag pa;
    const bf16_t* pp = ps + (m0 + l15) * 72 + kstep * 32 + half * 8;
    pa.q[0] = *(const uint4*)pp;
    pa.q[1] = *(const uint4*)(pp + 16);
    AFrag vb[2];
#pragma unroll
    for (int nt = 0; nt < 2; ++nt) {
      const bf16_t* vp = vts + (nt * 16 + l15) * 72 + kstep * 32 + half * 8;
      vb[nt].q[0] = *(const uint4*)vp;
      vb[nt].q[1] = *(const uint4*)(vp + 16);
    }
#pragma unroll
    for (int nt = 0; nt < 2; ++nt) {
      oacc[nt] = wmma_bf16(pa.v, vb[nt].v, oacc[nt]);
    }
  }
#pragma unroll
  for (int nt = 0; nt < 2; ++nt) {
#pragma unroll
    for (int r = 0; r < 8; ++r) {
      const int row = m0 + 8 * half + r;
      if (row < 49) {
        const int d = nt * 16 + l15;
        o[((size_t)win * 49 + row) * 256 + head * 32 + d] = (bf16_t)oacc[nt][r];
      }
    }
  }
}

// ---------------------------------------------------------------------------
extern "C" void kernel_launch(void* const* d_in, const int* in_sizes, int n_in,
                              void* d_out, int out_size, void* d_ws, size_t ws_size,
                              hipStream_t stream) {
  const float* x      = (const float*)d_in[0];
  const float* qg     = (const float*)d_in[1];
  const float* ln1_g  = (const float*)d_in[2];
  const float* ln1_b  = (const float*)d_in[3];
  const float* kv_w   = (const float*)d_in[4];
  const float* kv_b   = (const float*)d_in[5];
  const float* rpb    = (const float*)d_in[6];
  const float* proj_w = (const float*)d_in[7];
  const float* proj_b = (const float*)d_in[8];
  const float* ln2_g  = (const float*)d_in[9];
  const float* ln2_b  = (const float*)d_in[10];
  const float* fc1_w  = (const float*)d_in[11];
  const float* fc1_b  = (const float*)d_in[12];
  const float* fc2_w  = (const float*)d_in[13];
  const float* fc2_b  = (const float*)d_in[14];
  float* out = (float*)d_out;
  (void)in_sizes; (void)n_in; (void)out_size; (void)ws_size;

  const size_t M = 100352;  // = 32*56*56 = 2048 windows * 49 tokens
  char* w = (char*)d_ws;
  size_t off = 0;
  auto alloc = [&](size_t bytes) -> char* {
    char* p = w + off;
    off = (off + bytes + 255) & ~(size_t)255;
    return p;
  };
  bf16_t* xw    = (bf16_t*)alloc(M * 256 * 2);            // LN1(x), window order
  bf16_t* kvb   = (bf16_t*)alloc(M * 512 * 2);            // k|v, bf16
  bf16_t* qb    = (bf16_t*)alloc((size_t)32 * 8 * 49 * 32 * 2);
  float*  biasP = (float*) alloc((size_t)8 * 64 * 64 * 4);
  bf16_t* ao    = (bf16_t*)alloc(M * 256 * 2);            // attention output
  float*  xnew  = (float*) alloc(M * 256 * 4);            // x + attn (raster)
  bf16_t* xn2   = (bf16_t*)alloc(M * 256 * 2);            // LN2(xnew)
  bf16_t* hbuf  = (bf16_t*)alloc(M * 1024 * 2);           // gelu(fc1)
  bf16_t* wkv   = (bf16_t*)alloc((size_t)512 * 256 * 2);
  bf16_t* wproj = (bf16_t*)alloc((size_t)256 * 256 * 2);
  bf16_t* wfc1  = (bf16_t*)alloc((size_t)1024 * 256 * 2);
  bf16_t* wfc2  = (bf16_t*)alloc((size_t)256 * 1024 * 2);

  cvt_f32_bf16<<<(131072 + 255) / 256, 256, 0, stream>>>(kv_w, wkv, 131072);
  cvt_f32_bf16<<<(65536 + 255) / 256, 256, 0, stream>>>(proj_w, wproj, 65536);
  cvt_f32_bf16<<<(262144 + 255) / 256, 256, 0, stream>>>(fc1_w, wfc1, 262144);
  cvt_f32_bf16<<<(262144 + 255) / 256, 256, 0, stream>>>(fc2_w, wfc2, 262144);
  cvt_f32_bf16<<<(401408 + 255) / 256, 256, 0, stream>>>(qg, qb, 401408);
  build_bias_pad<<<(32768 + 255) / 256, 256, 0, stream>>>(rpb, biasP);

  // LN1 + window partition -> bf16
  ln_rows<<<12544, 256, 0, stream>>>(x, ln1_g, ln1_b, xw, 1);
  // kv = xw @ kv_w^T + kv_b
  gemm_bf16_ep<EPI_KV><<<dim3(784, 4), 256, 0, stream>>>(
      xw, wkv, kv_b, nullptr, nullptr, kvb, 256, 512);
  // windowed attention
  attn_win<<<16384, 128, 0, stream>>>(qb, kvb, biasP, ao);
  // proj + window reverse + residual -> xnew (f32)
  gemm_bf16_ep<EPI_PROJ><<<dim3(784, 2), 256, 0, stream>>>(
      ao, wproj, proj_b, x, xnew, nullptr, 256, 256);
  // LN2 -> bf16
  ln_rows<<<12544, 256, 0, stream>>>(xnew, ln2_g, ln2_b, xn2, 0);
  // fc1 + GELU -> bf16
  gemm_bf16_ep<EPI_FC1><<<dim3(784, 8), 256, 0, stream>>>(
      xn2, wfc1, fc1_b, nullptr, nullptr, hbuf, 256, 1024);
  // fc2 + residual -> out (f32)
  gemm_bf16_ep<EPI_FC2><<<dim3(784, 2), 256, 0, stream>>>(
      hbuf, wfc2, fc2_b, xnew, out, nullptr, 1024, 256);
}
